// NemotronFFN_MoE_43946105372961
// MI455X (gfx1250) — matrix operand order repaired and checked by reading, not compiled, not run on hardware
//
#include <hip/hip_runtime.h>

// ---------------------------------------------------------------------------
// NemotronFFN MoE for MI455X (gfx1250, wave32, WMMA bf16 16x16x32).
// Workspace layout (needs ~1.5 MB):
//   gate  : T*E  fp32   dense routing gates (0 if expert not selected)
//   tlist : E*T  int    per-expert gathered token ids
//   glist : E*T  fp32   per-expert gathered gate weights
//   counts: E    int    per-expert token counts
// ---------------------------------------------------------------------------

#define D_DIM   2048
#define E_N     32
#define T_TOT   4096
#define M_EXP   1024     // expert intermediate; shared expert = 2 slices of this
#define TOK_T   16       // token tile per block
#define WAVES   8
#define THREADS 256
#define SCALE_F 2.5f

typedef __bf16 bf16_t;
typedef bf16_t v16bf __attribute__((ext_vector_type(16)));
typedef bf16_t v8bf  __attribute__((ext_vector_type(8)));
typedef float  v8f   __attribute__((ext_vector_type(8)));
typedef float  v4f   __attribute__((ext_vector_type(4)));

__device__ __forceinline__ v8f zero_v8f() {
    v8f z;
#pragma unroll
    for (int i = 0; i < 8; ++i) z[i] = 0.0f;
    return z;
}

// B fragment from fp32 row: 8 floats at p, 8 floats at p+16 (K and K+16 halves)
__device__ __forceinline__ v16bf load16_cvt(const float* p) {
    const v4f* q = (const v4f*)p;
    v4f f0 = q[0], f1 = q[1], f2 = q[4], f3 = q[5];
    v16bf r;
#pragma unroll
    for (int i = 0; i < 4; ++i) {
        r[i]      = (bf16_t)f0[i];
        r[4 + i]  = (bf16_t)f1[i];
        r[8 + i]  = (bf16_t)f2[i];
        r[12 + i] = (bf16_t)f3[i];
    }
    return r;
}

// A fragment from fp32 row: 8 floats at p, 8 floats at p+16 (K and K+16 halves)
__device__ __forceinline__ v16bf loada_cvt(const float* p) {
    const v4f* q = (const v4f*)p;
    v4f f0 = q[0], f1 = q[1], f2 = q[4], f3 = q[5];
    v16bf r;
#pragma unroll
    for (int i = 0; i < 4; ++i) {
        r[i]      = (bf16_t)f0[i];
        r[4 + i]  = (bf16_t)f1[i];
        r[8 + i]  = (bf16_t)f2[i];
        r[12 + i] = (bf16_t)f3[i];
    }
    return r;
}

// A fragment from bf16 LDS row: 8 elems at p, 8 elems at p+16
__device__ __forceinline__ v16bf loada_lds(const bf16_t* p) {
    const v8bf* q = (const v8bf*)p;
    v8bf a0 = q[0], a1 = q[2];
    v16bf r;
#pragma unroll
    for (int i = 0; i < 8; ++i) { r[i] = a0[i]; r[8 + i] = a1[i]; }
    return r;
}

// ---------------------------------------------------------------------------
// Router: sigmoid(x @ Wr^T + br), grouped top-k, dense gate output.
// Block = 128 threads = 4 waves = 4 tokens; lane = expert.
// ---------------------------------------------------------------------------
__global__ __launch_bounds__(128)
void router_kernel(const float* __restrict__ x, const float* __restrict__ Wr,
                   const float* __restrict__ br, float* __restrict__ gate) {
    __shared__ float wrs[E_N][128 + 1];
    __shared__ float sbuf[4][E_N];
    __shared__ float gbuf[4][E_N];
    const int tid = threadIdx.x, wave = tid >> 5, lane = tid & 31;
    const int t = blockIdx.x * 4 + wave;

    float acc = 0.0f;
    for (int c = 0; c < D_DIM; c += 128) {
        __syncthreads();
        for (int i = tid; i < E_N * 128; i += 128)
            wrs[i >> 7][i & 127] = Wr[(size_t)(i >> 7) * D_DIM + c + (i & 127)];
        __syncthreads();
        const float* xp = x + (size_t)t * D_DIM + c;
#pragma unroll 8
        for (int j = 0; j < 128; ++j) acc += xp[j] * wrs[lane][j];
    }
    float s = 1.0f / (1.0f + __expf(-(acc + br[lane])));
    sbuf[wave][lane] = s;
    __syncthreads();

    if (lane == 0) {
        float sv[E_N];
#pragma unroll
        for (int e = 0; e < E_N; ++e) sv[e] = sbuf[wave][e];
        // per-group sum of top-2
        float gw[8];
        for (int g = 0; g < 8; ++g) {
            float m1 = -1.0f, m2 = -1.0f;
            for (int j = 0; j < 4; ++j) {
                float v = sv[g * 4 + j];
                if (v > m1) { m2 = m1; m1 = v; } else if (v > m2) { m2 = v; }
            }
            gw[g] = m1 + m2;
        }
        // top-4 groups
        bool gsel[8];
        for (int g = 0; g < 8; ++g) gsel[g] = false;
        for (int i = 0; i < 4; ++i) {
            int bi = 0; float best = -1e30f;
            for (int g = 0; g < 8; ++g)
                if (!gsel[g] && gw[g] > best) { best = gw[g]; bi = g; }
            gsel[bi] = true;
        }
        float ms[E_N];
        for (int e = 0; e < E_N; ++e) ms[e] = gsel[e >> 2] ? sv[e] : 0.0f;
        // top-6 experts; weights gathered from unmasked sigmoid scores
        int idx[6]; float wsum = 0.0f;
        for (int k = 0; k < 6; ++k) {
            int bi = 0; float best = -1.0f;
            for (int e = 0; e < E_N; ++e)
                if (ms[e] > best) { best = ms[e]; bi = e; }
            idx[k] = bi; ms[bi] = -2.0f; wsum += sv[bi];
        }
        const float inv = SCALE_F / (wsum + 1e-20f);
        for (int e = 0; e < E_N; ++e) gbuf[wave][e] = 0.0f;
        for (int k = 0; k < 6; ++k) gbuf[wave][idx[k]] = sv[idx[k]] * inv;
    }
    __syncthreads();
    gate[(size_t)t * E_N + lane] = gbuf[wave][lane];
}

// ---------------------------------------------------------------------------
// Deterministic order-preserving compaction: 1 wave per expert.
// ---------------------------------------------------------------------------
__global__ __launch_bounds__(32)
void compact_kernel(const float* __restrict__ gate, int* __restrict__ tlist,
                    float* __restrict__ glist, int* __restrict__ counts) {
    const int e = blockIdx.x, lane = threadIdx.x;
    int cnt = 0;
    for (int t0 = 0; t0 < T_TOT; t0 += 32) {
        const int t = t0 + lane;
        const float g = gate[(size_t)t * E_N + e];
        const unsigned mask = (unsigned)__ballot(g > 0.0f);
        const int prefix = __popc(mask & ((1u << lane) - 1u));
        if (g > 0.0f) {
            tlist[e * T_TOT + cnt + prefix] = t;
            glist[e * T_TOT + cnt + prefix] = g;
        }
        cnt += __popc(mask);
    }
    if (lane == 0) counts[e] = cnt;
}

// ---------------------------------------------------------------------------
// Fused FFN slice: out[tile] += gate * ( relu(x @ W1^T)^2 @ W2^T )
// One block = (slice e = blockIdx.y, token tile = blockIdx.x), 8 waves.
// GATHER: token ids/gates from per-expert lists; else identity tokens, gate=1.
// ---------------------------------------------------------------------------
template <bool GATHER>
__global__ __launch_bounds__(THREADS)
void ffn_kernel(const float* __restrict__ x,
                const float* __restrict__ W1, const float* __restrict__ W2,
                size_t w1stride, size_t w2stride, int ldb2,
                const int* __restrict__ tlist, const float* __restrict__ glist,
                const int* __restrict__ counts, float* __restrict__ out) {
    const int e = blockIdx.y;
    const int start = blockIdx.x * TOK_T;
    int nrows = TOK_T;
    if constexpr (GATHER) {
        const int cnt = counts[e];
        if (start >= cnt) return;                    // block-uniform exit
        nrows = (cnt - start < TOK_T) ? (cnt - start) : TOK_T;
    }
    const float* W1p = W1 + w1stride * (size_t)e;
    const float* W2p = W2 + w2stride * (size_t)e;

    __shared__ int   toks[TOK_T];
    __shared__ float gates[TOK_T];
    __shared__ bf16_t hbuf[TOK_T * (M_EXP + 8)];     // 16 x 1032 bf16 = 33 KB

    const int tid = threadIdx.x;
    if (tid < TOK_T) {
        if constexpr (GATHER) {
            const int src = start + ((tid < nrows) ? tid : 0);
            toks[tid]  = tlist[e * T_TOT + src];
            gates[tid] = (tid < nrows) ? glist[e * T_TOT + src] : 0.0f;
        } else {
            toks[tid]  = start + tid;
            gates[tid] = 1.0f;
        }
    }
    __syncthreads();

    const int wave = tid >> 5, lane = tid & 31;
    const int mrow = lane & 15;             // A-matrix row (token within tile)
    const int koff = (lane >> 4) * 8;       // A-matrix K sub-offset
    const int bn   = lane & 15;             // B-matrix column within tile
    const int bk   = (lane >> 4) * 8;       // B-matrix K sub-offset
    const int drow = (lane >> 4) * 8;       // D-matrix row base (M = v + drow)

    // ---- Phase 1: h = relu(x @ W1^T)^2  (bf16 into LDS) ------------------
    {
        constexpr int NT1 = M_EXP / 16 / WAVES;      // 8 column tiles / wave
        const float* arow = x + (size_t)toks[mrow] * D_DIM;
        v8f acc[NT1];
#pragma unroll
        for (int j = 0; j < NT1; ++j) acc[j] = zero_v8f();

        for (int kb = 0; kb < D_DIM; kb += 32) {
            const v16bf a = loada_cvt(arow + kb + koff);
#pragma unroll
            for (int j = 0; j < NT1; ++j) {
                const int nt = (wave * NT1 + j) * 16;
                const v16bf b =
                    load16_cvt(W1p + (size_t)(nt + bn) * D_DIM + kb + bk);
                acc[j] = __builtin_amdgcn_wmma_f32_16x16x32_bf16(
                    false, a, false, b, (short)0, acc[j], false, false);
            }
        }
#pragma unroll
        for (int j = 0; j < NT1; ++j) {
            const int nt = (wave * NT1 + j) * 16;
#pragma unroll
            for (int v = 0; v < 8; ++v) {
                float r = acc[j][v];
                r = (r > 0.0f) ? r : 0.0f;
                r = r * r;
                hbuf[(v + drow) * (M_EXP + 8) + nt + bn] = (bf16_t)r;
            }
        }
    }
    __syncthreads();

    // ---- Phase 2: out += gate * (h @ W2^T) -------------------------------
    {
        constexpr int NT2 = D_DIM / 16 / WAVES;      // 16 column tiles / wave
        constexpr int CH2 = 4;                       // tiles per register chunk
        const bf16_t* hrow = &hbuf[mrow * (M_EXP + 8)];

        for (int c = 0; c < NT2; c += CH2) {
            v8f acc[CH2];
#pragma unroll
            for (int j = 0; j < CH2; ++j) acc[j] = zero_v8f();

            for (int kb = 0; kb < M_EXP; kb += 32) {
                const v16bf a = loada_lds(hrow + kb + koff);
#pragma unroll
                for (int j = 0; j < CH2; ++j) {
                    const int nt = (wave * NT2 + c + j) * 16;
                    const v16bf b =
                        load16_cvt(W2p + (size_t)(nt + bn) * ldb2 + kb + bk);
                    acc[j] = __builtin_amdgcn_wmma_f32_16x16x32_bf16(
                        false, a, false, b, (short)0, acc[j], false, false);
                }
            }
#pragma unroll
            for (int j = 0; j < CH2; ++j) {
                const int nt = (wave * NT2 + c + j) * 16;
#pragma unroll
                for (int v = 0; v < 8; ++v) {
                    const int m = v + drow;
                    const float val = gates[m] * acc[j][v];
                    atomicAdd(out + (size_t)toks[m] * D_DIM + nt + bn, val);
                }
            }
        }
    }
}

// ---------------------------------------------------------------------------
extern "C" void kernel_launch(void* const* d_in, const int* in_sizes, int n_in,
                              void* d_out, int out_size, void* d_ws,
                              size_t ws_size, hipStream_t stream) {
    const float* x   = (const float*)d_in[0];   // [2,2048,2048]
    const float* Wr  = (const float*)d_in[1];   // [32,2048]
    const float* br  = (const float*)d_in[2];   // [32]
    const float* W1  = (const float*)d_in[3];   // [32,1024,2048]
    const float* W2  = (const float*)d_in[4];   // [32,2048,1024]
    const float* Ws1 = (const float*)d_in[5];   // [2048,2048]
    const float* Ws2 = (const float*)d_in[6];   // [2048,2048]
    float* out = (float*)d_out;

    char* ws = (char*)d_ws;
    float* gate   = (float*)ws;                               // T*E
    int*   tlist  = (int*)(ws + (size_t)T_TOT * E_N * 4);     // E*T
    float* glist  = (float*)(ws + (size_t)T_TOT * E_N * 8);   // E*T
    int*   counts = (int*)(ws + (size_t)T_TOT * E_N * 12);    // E

    hipMemsetAsync(d_out, 0, (size_t)out_size * sizeof(float), stream);

    router_kernel<<<T_TOT / 4, 128, 0, stream>>>(x, Wr, br, gate);
    compact_kernel<<<E_N, 32, 0, stream>>>(gate, tlist, glist, counts);

    // Shared expert as 2 slices of M=1024 (W2 slice = column offset, ld=2048)
    ffn_kernel<false><<<dim3(T_TOT / TOK_T, 2), THREADS, 0, stream>>>(
        x, Ws1, Ws2, (size_t)M_EXP * D_DIM, (size_t)M_EXP, D_DIM,
        nullptr, nullptr, nullptr, out);

    // Routed experts (gathered token tiles)
    ffn_kernel<true><<<dim3(T_TOT / TOK_T, E_N), THREADS, 0, stream>>>(
        x, W1, W2, (size_t)M_EXP * D_DIM, (size_t)D_DIM * M_EXP, M_EXP,
        tlist, glist, counts, out);
}
